// CMMD_loss_23630910063183
// MI455X (gfx1250) — matrix operand order repaired and checked by reading, not compiled, not run on hardware
//
#include <hip/hip_runtime.h>
#include <hip/hip_bf16.h>

typedef __attribute__((ext_vector_type(4)))  float  v4f;
typedef __attribute__((ext_vector_type(8)))  float  v8f;
typedef __attribute__((ext_vector_type(4)))  __bf16 v4bf;
typedef __attribute__((ext_vector_type(8)))  __bf16 v8bf;
typedef __attribute__((ext_vector_type(16))) __bf16 v16bf;

#define DIMD   2048
#define BSZ    2048   // rows in source (== rows in target)
#define NROW   4096
#define LDSTR  40     // padded LDS K-stride (bf16 elems): 80B, 16B-aligned, conflict-free

// CDNA5 LDS-only workgroup barrier: drain DScnt, then split signal/wait.
// Unlike __syncthreads(), does NOT drain LOADcnt, so in-flight global
// prefetch loads stay asynchronous across the barrier.
__device__ __forceinline__ void lds_barrier() {
    asm volatile("s_wait_dscnt 0x0\n\t"
                 "s_barrier_signal -1\n\t"
                 "s_barrier_wait -1" ::: "memory");
}

// ---------------------------------------------------------------------------
// K1: per-row sum of squares. One 256-thread block per row (4096 blocks).
// ---------------------------------------------------------------------------
__global__ __launch_bounds__(256) void cmmd_rowsq(const float* __restrict__ src,
                                                  const float* __restrict__ tgt,
                                                  float* __restrict__ sq) {
    __shared__ float red[256];
    const int row = blockIdx.x;
    const float* rp = (row < BSZ) ? (src + (size_t)row * DIMD)
                                  : (tgt + (size_t)(row - BSZ) * DIMD);
    float s = 0.f;
    for (int c = threadIdx.x; c < DIMD; c += 256) { float x = rp[c]; s += x * x; }
    red[threadIdx.x] = s;
    __syncthreads();
    for (int st = 128; st > 0; st >>= 1) {
        if (threadIdx.x < st) red[threadIdx.x] += red[threadIdx.x + st];
        __syncthreads();
    }
    if (threadIdx.x == 0) sq[row] = red[0];
}

// ---------------------------------------------------------------------------
// K2: column sums of the concatenated matrix (for sum(L2) identity).
// ---------------------------------------------------------------------------
__global__ __launch_bounds__(256) void cmmd_colsum(const float* __restrict__ src,
                                                   const float* __restrict__ tgt,
                                                   float* __restrict__ colsum) {
    const int d = blockIdx.x * 256 + threadIdx.x;
    if (d >= DIMD) return;
    double a = 0.0;
    for (int i = 0; i < BSZ; ++i) a += (double)src[(size_t)i * DIMD + d];
    for (int i = 0; i < BSZ; ++i) a += (double)tgt[(size_t)i * DIMD + d];
    colsum[d] = (float)a;
}

// ---------------------------------------------------------------------------
// K3: bandwidth.  sum(L2) = 2*N*sum(sq) - 2*||colsum||^2  (exact identity).
// Store scal = -1/(16*bw) so epilogue does e = exp(scal*L2); e,e^2,..,e^16.
// ---------------------------------------------------------------------------
__global__ __launch_bounds__(256) void cmmd_bw(const float* __restrict__ sq,
                                               const float* __restrict__ colsum,
                                               float* __restrict__ scal) {
    __shared__ double r1[256], r2[256];
    double a = 0.0, b = 0.0;
    for (int i = threadIdx.x; i < NROW; i += 256) a += (double)sq[i];
    for (int d = threadIdx.x; d < DIMD; d += 256) { double c = colsum[d]; b += c * c; }
    r1[threadIdx.x] = a; r2[threadIdx.x] = b;
    __syncthreads();
    for (int st = 128; st > 0; st >>= 1) {
        if (threadIdx.x < st) { r1[threadIdx.x] += r1[threadIdx.x + st];
                                r2[threadIdx.x] += r2[threadIdx.x + st]; }
        __syncthreads();
    }
    if (threadIdx.x == 0) {
        double sumL2 = 2.0 * (double)NROW * r1[0] - 2.0 * r2[0];
        double bw = sumL2 / ((double)NROW * (double)NROW - (double)NROW);
        bw *= 0.25;                       // / kernel_mul^(kernel_num//2) = /4
        scal[0] = (float)(-1.0 / (16.0 * bw));
    }
}

// ---------------------------------------------------------------------------
// K-prep: one-shot fp32 -> bf16 hi/lo split (removes 32x redundant conversion
// from the GEMM hot loop).  8192 blocks x 256 threads, 4 elems/thread.
// ---------------------------------------------------------------------------
__global__ __launch_bounds__(256) void cmmd_split(const float* __restrict__ src,
                                                  const float* __restrict__ tgt,
                                                  __bf16* __restrict__ Xhi,
                                                  __bf16* __restrict__ Xlo) {
    const size_t idx = ((size_t)blockIdx.x * 256 + threadIdx.x) * 4;
    const size_t half = (size_t)BSZ * DIMD;
    const float* p = (idx < half) ? (src + idx) : (tgt + (idx - half));
    v4f x = *(const v4f*)p;
    v4bf h, l;
#pragma unroll
    for (int e = 0; e < 4; ++e) {
        __bf16 hb = (__bf16)x[e];
        h[e] = hb;
        l[e] = (__bf16)(x[e] - (float)hb);
    }
    *(v4bf*)&Xhi[idx] = h;
    *(v4bf*)&Xlo[idx] = l;
}

// ---------------------------------------------------------------------------
// Shared epilogue: L2 -> 5 kernels via squaring chain -> masked +- sum,
// deterministic reduction into partial[block].
// ---------------------------------------------------------------------------
__device__ __forceinline__ void cmmd_epilogue(const v8f acc[4][2],
                                              int M0, int N0, int aBase, int bBase,
                                              int lrow, int lhi, int lane, int wave, int tid,
                                              const int* __restrict__ slab,
                                              const int* __restrict__ tlab,
                                              const float* __restrict__ sq,
                                              const float* __restrict__ scal,
                                              float* __restrict__ partial,
                                              float* wred, int blk) {
    const float s0 = scal[0];   // = -1/(16*bw)
    float psum = 0.f;
#pragma unroll
    for (int nt = 0; nt < 2; ++nt) {
        const int j  = N0 + bBase + nt * 16 + lrow;
        const int lj = (j < BSZ) ? slab[j] : tlab[j - BSZ];
        const bool jsrc = (j < BSZ);
        const float sqj = sq[j];
#pragma unroll
        for (int mt = 0; mt < 4; ++mt) {
#pragma unroll
            for (int v = 0; v < 8; ++v) {
                const int i  = M0 + aBase + mt * 16 + v + lhi * 8;
                const int li = (i < BSZ) ? slab[i] : tlab[i - BSZ];
                if (li == lj) {
                    const float L2 = sq[i] + sqj - 2.0f * acc[mt][nt][v];
                    const float e1 = expf(s0 * L2);
                    const float e2 = e1 * e1, e4 = e2 * e2, e8 = e4 * e4, e16 = e8 * e8;
                    const float val = e1 + e2 + e4 + e8 + e16;
                    psum += (((i < BSZ) == jsrc) ? val : -val);
                }
            }
        }
    }
    for (int off = 16; off > 0; off >>= 1) psum += __shfl_down(psum, off);
    if (lane == 0) wred[wave] = psum;
    __syncthreads();
    if (tid == 0) {
        float t = 0.f;
        for (int w = 0; w < 8; ++w) t += wred[w];
        partial[blk] = t;
    }
}

// ---------------------------------------------------------------------------
// K4-fast: double-buffered LDS GEMM from precomputed bf16 hi/lo.
// 128x128 tile / 256-thread block; wave = 64x32 (4x2 WMMA tiles, 24 wmma/chunk).
// One LDS-only barrier per chunk; next chunk's global loads stay in flight
// across the barrier and are waited only at the ds_store at the block's end.
// ---------------------------------------------------------------------------
__global__ __launch_bounds__(256) void cmmd_gram_pre(const __bf16* __restrict__ Xhi,
                                                     const __bf16* __restrict__ Xlo,
                                                     const int* __restrict__ slab,
                                                     const int* __restrict__ tlab,
                                                     const float* __restrict__ sq,
                                                     const float* __restrict__ scal,
                                                     float* __restrict__ partial) {
    __shared__ __bf16 sA[2][2][128 * LDSTR];   // [buf][hi/lo]
    __shared__ __bf16 sB[2][2][128 * LDSTR];
    __shared__ float  wred[8];

    const int tile = blockIdx.x;
    const int M0 = (tile >> 5) * 128;
    const int N0 = (tile & 31) * 128;

    const int tid  = threadIdx.x;
    const int lane = tid & 31;
    const int wave = tid >> 5;
    const int lrow = lane & 15;
    const int lhi  = lane >> 4;
    const int aBase = (wave & 1) * 64;
    const int bBase = (wave >> 1) * 32;

    // staging mapping: thread -> (row 0..127, K-half 0/16)
    const int sr = tid >> 1;
    const int sh = (tid & 1) * 16;
    const size_t aoff = (size_t)(M0 + sr) * DIMD + sh;
    const size_t boff = (size_t)(N0 + sr) * DIMD + sh;
    const int so = sr * LDSTR + sh;

    v8bf rAh[2], rAl[2], rBh[2], rBl[2];
    auto gload = [&](int k0) {
        rAh[0] = *(const v8bf*)(Xhi + aoff + k0);
        rAh[1] = *(const v8bf*)(Xhi + aoff + k0 + 8);
        rAl[0] = *(const v8bf*)(Xlo + aoff + k0);
        rAl[1] = *(const v8bf*)(Xlo + aoff + k0 + 8);
        rBh[0] = *(const v8bf*)(Xhi + boff + k0);
        rBh[1] = *(const v8bf*)(Xhi + boff + k0 + 8);
        rBl[0] = *(const v8bf*)(Xlo + boff + k0);
        rBl[1] = *(const v8bf*)(Xlo + boff + k0 + 8);
    };
    auto lstore = [&](int buf) {
        *(v8bf*)&sA[buf][0][so]     = rAh[0];  *(v8bf*)&sA[buf][0][so + 8] = rAh[1];
        *(v8bf*)&sA[buf][1][so]     = rAl[0];  *(v8bf*)&sA[buf][1][so + 8] = rAl[1];
        *(v8bf*)&sB[buf][0][so]     = rBh[0];  *(v8bf*)&sB[buf][0][so + 8] = rBh[1];
        *(v8bf*)&sB[buf][1][so]     = rBl[0];  *(v8bf*)&sB[buf][1][so + 8] = rBl[1];
    };

    const v8f vzero = {0.f, 0.f, 0.f, 0.f, 0.f, 0.f, 0.f, 0.f};
    v8f acc[4][2];
#pragma unroll
    for (int mt = 0; mt < 4; ++mt)
#pragma unroll
        for (int nt = 0; nt < 2; ++nt) acc[mt][nt] = vzero;

    gload(0);
    lstore(0);

    for (int kc = 0; kc < DIMD / 32; ++kc) {
        const int cur = kc & 1;
        if (kc + 1 < DIMD / 32) gload((kc + 1) * 32);   // prefetch next chunk
        lds_barrier();                                   // DS-only: prefetch stays async

        v16bf aH[4], aL[4];
#pragma unroll
        for (int mt = 0; mt < 4; ++mt) {
            const int base = (aBase + mt * 16 + lrow) * LDSTR + lhi * 8;
            v8bf x0 = *(const v8bf*)&sA[cur][0][base];
            v8bf x1 = *(const v8bf*)&sA[cur][0][base + 16];
            aH[mt] = __builtin_shufflevector(x0, x1, 0,1,2,3,4,5,6,7,8,9,10,11,12,13,14,15);
            v8bf y0 = *(const v8bf*)&sA[cur][1][base];
            v8bf y1 = *(const v8bf*)&sA[cur][1][base + 16];
            aL[mt] = __builtin_shufflevector(y0, y1, 0,1,2,3,4,5,6,7,8,9,10,11,12,13,14,15);
        }
        v16bf bH[2], bL[2];
#pragma unroll
        for (int nt = 0; nt < 2; ++nt) {
            const int base = (bBase + nt * 16 + lrow) * LDSTR + lhi * 8;
            v8bf x0 = *(const v8bf*)&sB[cur][0][base];
            v8bf x1 = *(const v8bf*)&sB[cur][0][base + 16];
            bH[nt] = __builtin_shufflevector(x0, x1, 0,1,2,3,4,5,6,7,8,9,10,11,12,13,14,15);
            v8bf y0 = *(const v8bf*)&sB[cur][1][base];
            v8bf y1 = *(const v8bf*)&sB[cur][1][base + 16];
            bL[nt] = __builtin_shufflevector(y0, y1, 0,1,2,3,4,5,6,7,8,9,10,11,12,13,14,15);
        }

#pragma unroll
        for (int mt = 0; mt < 4; ++mt)
#pragma unroll
            for (int nt = 0; nt < 2; ++nt) {
                acc[mt][nt] = __builtin_amdgcn_wmma_f32_16x16x32_bf16(
                    false, aH[mt], false, bH[nt], (short)0, acc[mt][nt], false, false);
                acc[mt][nt] = __builtin_amdgcn_wmma_f32_16x16x32_bf16(
                    false, aH[mt], false, bL[nt], (short)0, acc[mt][nt], false, false);
                acc[mt][nt] = __builtin_amdgcn_wmma_f32_16x16x32_bf16(
                    false, aL[mt], false, bH[nt], (short)0, acc[mt][nt], false, false);
            }

        if (kc + 1 < DIMD / 32) lstore(1 - cur);   // safe: buffer last read 1 iter ago, before this iter's barrier
    }

    __syncthreads();   // full barrier before reduction reuses LDS
    cmmd_epilogue(acc, M0, N0, aBase, bBase, lrow, lhi, lane, wave, tid,
                  slab, tlab, sq, scal, partial, wred, blockIdx.x);
}

// ---------------------------------------------------------------------------
// K4-fallback: convert-on-the-fly variant (used only if ws is too small for
// the bf16 split arrays).  Same math, single-buffered.
// ---------------------------------------------------------------------------
__global__ __launch_bounds__(256) void cmmd_gram(const float* __restrict__ src,
                                                 const float* __restrict__ tgt,
                                                 const int* __restrict__ slab,
                                                 const int* __restrict__ tlab,
                                                 const float* __restrict__ sq,
                                                 const float* __restrict__ scal,
                                                 float* __restrict__ partial) {
    __shared__ __bf16 sAhi[128 * LDSTR];
    __shared__ __bf16 sAlo[128 * LDSTR];
    __shared__ __bf16 sBhi[128 * LDSTR];
    __shared__ __bf16 sBlo[128 * LDSTR];
    __shared__ float  wred[8];

    const int tile = blockIdx.x;
    const int M0 = (tile >> 5) * 128;
    const int N0 = (tile & 31) * 128;

    const int tid  = threadIdx.x;
    const int lane = tid & 31;
    const int wave = tid >> 5;
    const int lrow = lane & 15;
    const int lhi  = lane >> 4;
    const int aBase = (wave & 1) * 64;
    const int bBase = (wave >> 1) * 32;

    const int sr = tid >> 1;
    const int sh = (tid & 1) * 16;

    const int arow = M0 + sr;
    const float* aptr = (arow < BSZ) ? (src + (size_t)arow * DIMD)
                                     : (tgt + (size_t)(arow - BSZ) * DIMD);
    const int brow = N0 + sr;
    const float* bptr = (brow < BSZ) ? (src + (size_t)brow * DIMD)
                                     : (tgt + (size_t)(brow - BSZ) * DIMD);

    const v8f vzero = {0.f, 0.f, 0.f, 0.f, 0.f, 0.f, 0.f, 0.f};
    v8f acc[4][2];
#pragma unroll
    for (int mt = 0; mt < 4; ++mt)
#pragma unroll
        for (int nt = 0; nt < 2; ++nt) acc[mt][nt] = vzero;

    for (int k0 = 0; k0 < DIMD; k0 += 32) {
        lds_barrier();
        {
            v4f fa[4], fb[4];
#pragma unroll
            for (int q = 0; q < 4; ++q) fa[q] = *(const v4f*)(aptr + k0 + sh + 4 * q);
#pragma unroll
            for (int q = 0; q < 4; ++q) fb[q] = *(const v4f*)(bptr + k0 + sh + 4 * q);
            v8bf ah[2], al[2], bh[2], bl[2];
#pragma unroll
            for (int e = 0; e < 16; ++e) {
                float xa = fa[e >> 2][e & 3];
                __bf16 ha = (__bf16)xa;
                ah[e >> 3][e & 7] = ha;
                al[e >> 3][e & 7] = (__bf16)(xa - (float)ha);
                float xb = fb[e >> 2][e & 3];
                __bf16 hb = (__bf16)xb;
                bh[e >> 3][e & 7] = hb;
                bl[e >> 3][e & 7] = (__bf16)(xb - (float)hb);
            }
            const int so = sr * LDSTR + sh;
            *(v8bf*)&sAhi[so]     = ah[0];  *(v8bf*)&sAhi[so + 8] = ah[1];
            *(v8bf*)&sAlo[so]     = al[0];  *(v8bf*)&sAlo[so + 8] = al[1];
            *(v8bf*)&sBhi[so]     = bh[0];  *(v8bf*)&sBhi[so + 8] = bh[1];
            *(v8bf*)&sBlo[so]     = bl[0];  *(v8bf*)&sBlo[so + 8] = bl[1];
        }
        lds_barrier();

        v16bf aH[4], aL[4];
#pragma unroll
        for (int mt = 0; mt < 4; ++mt) {
            const int base = (aBase + mt * 16 + lrow) * LDSTR + lhi * 8;
            v8bf x0 = *(const v8bf*)&sAhi[base];
            v8bf x1 = *(const v8bf*)&sAhi[base + 16];
            aH[mt] = __builtin_shufflevector(x0, x1, 0,1,2,3,4,5,6,7,8,9,10,11,12,13,14,15);
            v8bf y0 = *(const v8bf*)&sAlo[base];
            v8bf y1 = *(const v8bf*)&sAlo[base + 16];
            aL[mt] = __builtin_shufflevector(y0, y1, 0,1,2,3,4,5,6,7,8,9,10,11,12,13,14,15);
        }
        v16bf bH[2], bL[2];
#pragma unroll
        for (int nt = 0; nt < 2; ++nt) {
            const int base = (bBase + nt * 16 + lrow) * LDSTR + lhi * 8;
            v8bf x0 = *(const v8bf*)&sBhi[base];
            v8bf x1 = *(const v8bf*)&sBhi[base + 16];
            bH[nt] = __builtin_shufflevector(x0, x1, 0,1,2,3,4,5,6,7,8,9,10,11,12,13,14,15);
            v8bf y0 = *(const v8bf*)&sBlo[base];
            v8bf y1 = *(const v8bf*)&sBlo[base + 16];
            bL[nt] = __builtin_shufflevector(y0, y1, 0,1,2,3,4,5,6,7,8,9,10,11,12,13,14,15);
        }

#pragma unroll
        for (int mt = 0; mt < 4; ++mt)
#pragma unroll
            for (int nt = 0; nt < 2; ++nt) {
                acc[mt][nt] = __builtin_amdgcn_wmma_f32_16x16x32_bf16(
                    false, aH[mt], false, bH[nt], (short)0, acc[mt][nt], false, false);
                acc[mt][nt] = __builtin_amdgcn_wmma_f32_16x16x32_bf16(
                    false, aH[mt], false, bL[nt], (short)0, acc[mt][nt], false, false);
                acc[mt][nt] = __builtin_amdgcn_wmma_f32_16x16x32_bf16(
                    false, aL[mt], false, bH[nt], (short)0, acc[mt][nt], false, false);
            }
    }

    __syncthreads();
    cmmd_epilogue(acc, M0, N0, aBase, bBase, lrow, lhi, lane, wave, tid,
                  slab, tlab, sq, scal, partial, wred, blockIdx.x);
}

// ---------------------------------------------------------------------------
// K5: final deterministic reduction of per-workgroup partials -> loss scalar.
// ---------------------------------------------------------------------------
__global__ __launch_bounds__(256) void cmmd_final(const float* __restrict__ partial,
                                                  float* __restrict__ out) {
    __shared__ double red[256];
    double a = 0.0;
    for (int i = threadIdx.x; i < 1024; i += 256) a += (double)partial[i];
    red[threadIdx.x] = a;
    __syncthreads();
    for (int st = 128; st > 0; st >>= 1) {
        if (threadIdx.x < st) red[threadIdx.x] += red[threadIdx.x + st];
        __syncthreads();
    }
    if (threadIdx.x == 0)
        out[0] = (float)(red[0] / ((double)BSZ * (double)BSZ));
}

// ---------------------------------------------------------------------------
extern "C" void kernel_launch(void* const* d_in, const int* in_sizes, int n_in,
                              void* d_out, int out_size, void* d_ws, size_t ws_size,
                              hipStream_t stream) {
    (void)in_sizes; (void)n_in; (void)out_size;
    const float* src = (const float*)d_in[0];
    const float* tgt = (const float*)d_in[1];
    const int*   sl  = (const int*)d_in[2];
    const int*   tl  = (const int*)d_in[3];
    float* out = (float*)d_out;

    // scratch layout: small arrays first (shared by both paths), then bf16 split
    float* sq      = (float*)d_ws;          // 4096
    float* colsum  = sq + NROW;             // 2048
    float* scal    = colsum + DIMD;         // 4
    float* partial = scal + 4;              // 1024
    const size_t smallBytes = 32768;        // 28.7 KB rounded up, keeps 16B align
    const size_t nElem = (size_t)NROW * DIMD;
    __bf16* Xhi = (__bf16*)((char*)d_ws + smallBytes);
    __bf16* Xlo = Xhi + nElem;
    const size_t need = smallBytes + nElem * 2 * sizeof(__bf16);

    cmmd_rowsq <<<NROW,       256, 0, stream>>>(src, tgt, sq);
    cmmd_colsum<<<DIMD / 256, 256, 0, stream>>>(src, tgt, colsum);
    cmmd_bw    <<<1,          256, 0, stream>>>(sq, colsum, scal);

    if (ws_size >= need) {
        cmmd_split   <<<(int)(nElem / 1024), 256, 0, stream>>>(src, tgt, Xhi, Xlo);
        cmmd_gram_pre<<<1024, 256, 0, stream>>>(Xhi, Xlo, sl, tl, sq, scal, partial);
    } else {
        cmmd_gram    <<<1024, 256, 0, stream>>>(src, tgt, sl, tl, sq, scal, partial);
    }
    cmmd_final <<<1, 256, 0, stream>>>(partial, out);
}
